// AutoEncoderCov3DMem_25701084299677
// MI455X (gfx1250) — matrix operand order repaired and checked by reading, not compiled, not run on hardware
//
#include <hip/hip_runtime.h>

typedef unsigned short u16;
typedef unsigned int   u32;
typedef long long      i64;

typedef __attribute__((ext_vector_type(16))) __bf16 v16bf;
typedef __attribute__((ext_vector_type(8)))  float  v8f;

union ABf  { v16bf v; u32 u[8]; };
union ACCf { v8f v; float f[8]; };

__device__ __forceinline__ u16 f2bf(float f) {
  u32 u = __float_as_uint(f);
  u += 0x7FFFu + ((u >> 16) & 1u);           // round-to-nearest-even
  return (u16)(u >> 16);
}

// Async global->LDS copy (CDNA5 async path, tracked by ASYNCcnt).
// low 32 bits of a flat shared pointer are the
// wave-relative LDS offset (aperture mapping, ISA 10.2).
__device__ __forceinline__ void async_load_b32(u32 ldsOff, const void* gaddr) {
  asm volatile("global_load_async_to_lds_b32 %0, %1, off"
               :: "v"(ldsOff), "v"(gaddr) : "memory");
}

// ---------------------------------------------------------------------------
__global__ void k_f32_to_bf16(const float* __restrict__ in, u16* __restrict__ out, i64 n) {
  for (i64 i = blockIdx.x * (i64)blockDim.x + threadIdx.x; i < n;
       i += (i64)gridDim.x * blockDim.x)
    out[i] = f2bf(in[i]);
}

__global__ void k_zero(float* __restrict__ p, int n) {
  int i = blockIdx.x * blockDim.x + threadIdx.x;
  if (i < n) p[i] = 0.f;
}

// ---------------------------------------------------------------------------
// Weight reorder: W[Cout][Cin][3][3][3] f32 -> WMMA-B fragment-major bf16 pairs:
//   wf[((ntile*nK + ks)*32 + lane)*8 + p] = pair(k, n)
//   k = ks*32 + (p>>2)*16 + (lane>>4)*8 + 2(p&3) (+h),  n = ntile*16 + (lane&15)
// K index = tap*Cin + ci, tap = kd*9+kh*3+kw (flipped for convT).
// ---------------------------------------------------------------------------
__global__ void k_reorder_w_frag(const float* __restrict__ w, u32* __restrict__ wf,
                                 int Cout, int Cin, int CoutPad, int Kpad, int flip) {
  const int Ktot = 27 * Cin;
  const int nK = Kpad >> 5;
  const i64 total = (i64)(CoutPad >> 4) * nK * 256;
  for (i64 idx = blockIdx.x * (i64)blockDim.x + threadIdx.x; idx < total;
       idx += (i64)gridDim.x * blockDim.x) {
    const int p    = (int)(idx & 7);
    const int lane = (int)((idx >> 3) & 31);
    const i64 r    = idx >> 8;
    const int ks    = (int)(r % nK);
    const int ntile = (int)(r / nK);
    const int n     = ntile * 16 + (lane & 15);
    const int kbase = ks * 32 + (p >> 2) * 16 + (lane >> 4) * 8 + 2 * (p & 3);
    u32 v = 0;
#pragma unroll
    for (int h = 0; h < 2; ++h) {
      const int k = kbase + h;
      if (k < Ktot && n < Cout) {
        const int tap = k / Cin, ci = k % Cin;
        int kd = tap / 9, kh = (tap / 3) % 3, kw = tap % 3;
        if (flip) { kd = 2 - kd; kh = 2 - kh; kw = 2 - kw; }
        const float f = w[((i64)n * Cin + ci) * 27 + kd * 9 + kh * 3 + kw];
        v |= ((u32)f2bf(f)) << (16 * h);
      }
    }
    wf[idx] = v;
  }
}

// ---------------------------------------------------------------------------
// Implicit-GEMM conv / convT with WMMA bf16 (f32 accumulate).
// Activations NDHWC bf16.  Per-dim mapping: t = o*stride + k - 1; valid iff
// t % dil == 0 and 0 <= t/dil < Din.  Forward: dil=1.  ConvT: stride=1,
// dil=upsample, weights pre-flipped.
// Block = 32 M-rows x 128 N-cols; wave computes two 16x16 tiles (rows 0-15 and
// 16-31) sharing one B fragment.  A tiles (32x32 K) staged in LDS via async
// global->LDS b32 DMA, double buffered; invalid taps read a zero page.
// ---------------------------------------------------------------------------
__global__ __launch_bounds__(256) void k_conv_wmma(
    const u16* __restrict__ act, const u32* __restrict__ wfrag,
    const float* __restrict__ bias, float* __restrict__ out,
    const u16* __restrict__ zeropad,
    int Din, int Hin, int Win, int Cin,
    int Dout, int Hout, int Wout, int CoutPad, int nStore,
    int sd, int sh, int sw, int dd, int dh, int dw,
    int Ktot, int Kpad) {
  __shared__ u32 Atile[2][512];            // [buf][32 rows x 16 K-pairs]
  const int tid  = threadIdx.x;
  const int wave = tid >> 5;
  const int lane = tid & 31;
  const int hiK  = lane >> 4;
  const int l16  = lane & 15;
  const u32 m0 = (u32)blockIdx.x * 32;
  const int n0 = blockIdx.y * 128 + wave * 16;
  const bool active = (n0 < CoutPad);
  const int nK = Kpad >> 5;
  const u32* wfw = wfrag + ((i64)(n0 >> 4) * nK) * 256 + lane * 8;

  // loader role: thread -> (rows lrow, lrow+16; K-pair lp)
  const int lp = tid & 15;
  const u32 HWout  = (u32)Hout * Wout;
  const u32 DHWout = (u32)Dout * HWout;
  int od0, oh0, ow0, nb0, od1, oh1, ow1, nb1;
  {
    u32 mm = m0 + (u32)(tid >> 4);
    u32 r = mm % DHWout; nb0 = (int)(mm / DHWout);
    od0 = (int)(r / HWout); r -= (u32)od0 * HWout;
    oh0 = (int)(r / (u32)Wout); ow0 = (int)(r - (u32)oh0 * Wout);
    mm += 16;
    r = mm % DHWout; nb1 = (int)(mm / DHWout);
    od1 = (int)(r / HWout); r -= (u32)od1 * HWout;
    oh1 = (int)(r / (u32)Wout); ow1 = (int)(r - (u32)oh1 * Wout);
  }

  const u32 ldsOff0  = (u32)(size_t)&Atile[0][tid];        // row lrow
  const u32 ldsOff0b = ldsOff0 + 256 * 4;                  // row lrow+16
  const u32 ldsOff1  = (u32)(size_t)&Atile[1][tid];
  const u32 ldsOff1b = ldsOff1 + 256 * 4;

  ACCf acc0, acc1;
#pragma unroll
  for (int i = 0; i < 8; ++i) { acc0.f[i] = 0.f; acc1.f[i] = 0.f; }

  auto rowBase = [&](int tap, int od, int oh, int ow, int nb) -> const u16* {
    const int kd = tap / 9, kh = (tap / 3) % 3, kw = tap % 3;
    const int td = od * sd + kd - 1;
    const int th = oh * sh + kh - 1;
    const int tw = ow * sw + kw - 1;
    if ((td % dd) != 0 || (th % dh) != 0 || (tw % dw) != 0) return nullptr;
    const int id = td / dd, ih = th / dh, iw = tw / dw;
    if ((unsigned)id >= (unsigned)Din || (unsigned)ih >= (unsigned)Hin ||
        (unsigned)iw >= (unsigned)Win) return nullptr;
    return act + ((((i64)nb * Din + id) * Hin + ih) * (i64)Win + iw) * Cin;
  };

  if (Cin == 1) {
    // layer 1: K = 27 padded to 32, single step, synchronous gather
    u32 av0 = 0u, av1 = 0u;
#pragma unroll
    for (int h = 0; h < 2; ++h) {
      const int kk = 2 * lp + h;
      if (kk < Ktot) {
        const u16* rb0 = rowBase(kk, od0, oh0, ow0, nb0);
        const u16* rb1 = rowBase(kk, od1, oh1, ow1, nb1);
        if (rb0) av0 |= ((u32)rb0[0]) << (16 * h);
        if (rb1) av1 |= ((u32)rb1[0]) << (16 * h);
      }
    }
    Atile[0][tid] = av0;
    Atile[0][tid + 256] = av1;
    __syncthreads();
    if (active) {
      ABf A0, A1, B;
      uint4 a0 = *(const uint4*)&Atile[0][l16 * 16 + hiK * 4];
      uint4 a1 = *(const uint4*)&Atile[0][l16 * 16 + 8 + hiK * 4];
      A0.u[0] = a0.x; A0.u[1] = a0.y; A0.u[2] = a0.z; A0.u[3] = a0.w;
      A0.u[4] = a1.x; A0.u[5] = a1.y; A0.u[6] = a1.z; A0.u[7] = a1.w;
      uint4 a2 = *(const uint4*)&Atile[0][256 + l16 * 16 + hiK * 4];
      uint4 a3 = *(const uint4*)&Atile[0][256 + l16 * 16 + 8 + hiK * 4];
      A1.u[0] = a2.x; A1.u[1] = a2.y; A1.u[2] = a2.z; A1.u[3] = a2.w;
      A1.u[4] = a3.x; A1.u[5] = a3.y; A1.u[6] = a3.z; A1.u[7] = a3.w;
      uint4 b0 = *(const uint4*)(wfw);
      uint4 b1 = *(const uint4*)(wfw + 4);
      B.u[0] = b0.x; B.u[1] = b0.y; B.u[2] = b0.z; B.u[3] = b0.w;
      B.u[4] = b1.x; B.u[5] = b1.y; B.u[6] = b1.z; B.u[7] = b1.w;
      acc0.v = __builtin_amdgcn_wmma_f32_16x16x32_bf16(
          false, A0.v, false, B.v, (short)0, acc0.v, false, false);
      acc1.v = __builtin_amdgcn_wmma_f32_16x16x32_bf16(
          false, A1.v, false, B.v, (short)0, acc1.v, false, false);
    }
  } else {
    // tap-major K: every 32-wide step lies inside one tap (Cin multiple of 32)
    int itap = 0, icb = 0;
    const u16 *base0 = nullptr, *base1 = nullptr;
    auto issueTile = [&](u32 offA, u32 offB) {
      if (icb == 0) {
        base0 = rowBase(itap, od0, oh0, ow0, nb0);
        base1 = rowBase(itap, od1, oh1, ow1, nb1);
      }
      const int co = icb + 2 * lp;
      async_load_b32(offA, base0 ? (const void*)(base0 + co) : (const void*)zeropad);
      async_load_b32(offB, base1 ? (const void*)(base1 + co) : (const void*)zeropad);
      icb += 32;
      if (icb >= Cin) { icb = 0; ++itap; }
    };

    issueTile(ldsOff0, ldsOff0b);
    for (int s = 0; s < nK; ++s) {
      const int cur = s & 1;
      if (s + 1 < nK) {
        issueTile(cur ? ldsOff0 : ldsOff1, cur ? ldsOff0b : ldsOff1b);
        asm volatile("s_wait_asynccnt 0x2" ::: "memory");   // tile s landed
      } else {
        asm volatile("s_wait_asynccnt 0x0" ::: "memory");
      }
      __syncthreads();
      if (active) {
        ABf A0, A1, B;
        uint4 a0 = *(const uint4*)&Atile[cur][l16 * 16 + hiK * 4];
        uint4 a1 = *(const uint4*)&Atile[cur][l16 * 16 + 8 + hiK * 4];
        A0.u[0] = a0.x; A0.u[1] = a0.y; A0.u[2] = a0.z; A0.u[3] = a0.w;
        A0.u[4] = a1.x; A0.u[5] = a1.y; A0.u[6] = a1.z; A0.u[7] = a1.w;
        uint4 a2 = *(const uint4*)&Atile[cur][256 + l16 * 16 + hiK * 4];
        uint4 a3 = *(const uint4*)&Atile[cur][256 + l16 * 16 + 8 + hiK * 4];
        A1.u[0] = a2.x; A1.u[1] = a2.y; A1.u[2] = a2.z; A1.u[3] = a2.w;
        A1.u[4] = a3.x; A1.u[5] = a3.y; A1.u[6] = a3.z; A1.u[7] = a3.w;
        const u32* bp = wfw + (i64)s * 256;
        uint4 b0 = *(const uint4*)(bp);
        uint4 b1 = *(const uint4*)(bp + 4);
        B.u[0] = b0.x; B.u[1] = b0.y; B.u[2] = b0.z; B.u[3] = b0.w;
        B.u[4] = b1.x; B.u[5] = b1.y; B.u[6] = b1.z; B.u[7] = b1.w;
        acc0.v = __builtin_amdgcn_wmma_f32_16x16x32_bf16(
            false, A0.v, false, B.v, (short)0, acc0.v, false, false);
        acc1.v = __builtin_amdgcn_wmma_f32_16x16x32_bf16(
            false, A1.v, false, B.v, (short)0, acc1.v, false, false);
      }
      __syncthreads();
    }
  }

  if (active) {
    const int col = n0 + l16;
    if (col < nStore) {
      const float bv = bias[col];
      const i64 r0 = (i64)m0 + hiK * 8;         // rows 0-15 tile
      const i64 r1 = r0 + 16;                   // rows 16-31 tile
#pragma unroll
      for (int r = 0; r < 8; ++r) {
        out[(r0 + r) * (i64)nStore + col] = acc0.f[r] + bv;
        out[(r1 + r) * (i64)nStore + col] = acc1.f[r] + bv;
      }
    }
  }
}

// ---------------------------------------------------------------------------
// Dense GEMM  C[M][N] f32 = A[M][K] bf16 x Bpack[K/2][N] (bf16 pairs).
// ---------------------------------------------------------------------------
__global__ __launch_bounds__(256) void k_gemm_wmma(
    const u16* __restrict__ A, const u32* __restrict__ Bp,
    float* __restrict__ C, int N, int K) {
  const int tid = threadIdx.x, wave = tid >> 5, lane = tid & 31;
  const int l16 = lane & 15, hiK = lane >> 4;
  const i64 m0 = (i64)blockIdx.x * 16;
  const int n0 = blockIdx.y * 128 + wave * 16;
  if (n0 >= N) return;
  const u32* arow = (const u32*)(A + (m0 + l16) * (i64)K);
  ACCf acc;
#pragma unroll
  for (int i = 0; i < 8; ++i) acc.f[i] = 0.f;
  for (int kb = 0; kb < K; kb += 32) {
    ABf Af, Bf;
    const int kp = kb >> 1;
#pragma unroll
    for (int p = 0; p < 4; ++p) {
      Af.u[p]     = arow[kp + hiK * 8 + p];
      Af.u[p + 4] = arow[kp + hiK * 8 + 4 + p];
    }
    const u32* bp = Bp + (i64)(kp + hiK * 8) * N + (n0 + l16);
#pragma unroll
    for (int p = 0; p < 8; ++p) Bf.u[p] = bp[(i64)p * N];
    acc.v = __builtin_amdgcn_wmma_f32_16x16x32_bf16(
        false, Af.v, false, Bf.v, (short)0, acc.v, false, false);
  }
  const int col = n0 + l16;
  const i64 rbase = m0 + hiK * 8;
#pragma unroll
  for (int r = 0; r < 8; ++r) C[(rbase + r) * (i64)N + col] = acc.f[r];
}

// ---------------------------------------------------------------------------
// BatchNorm (training-mode batch stats) over [M][C]
// ---------------------------------------------------------------------------
__global__ void k_bn_stats(const float* __restrict__ buf, float* __restrict__ stats,
                           i64 M, int C) {
  const int c = threadIdx.x;
  if (c >= C) return;
  float s = 0.f, q = 0.f;
  for (i64 m = blockIdx.x; m < M; m += gridDim.x) {
    const float v = buf[m * C + c];
    s += v; q += v * v;
  }
  atomicAdd(&stats[c], s);
  atomicAdd(&stats[C + c], q);
}

__global__ void k_bn_apply(const float* __restrict__ buf, const float* __restrict__ stats,
                           const float* __restrict__ g, const float* __restrict__ be,
                           u16* __restrict__ obf, float* __restrict__ of32,
                           i64 M, int C) {
  const i64 total = M * C;
  const float invM = 1.0f / (float)M;
  for (i64 idx = blockIdx.x * (i64)blockDim.x + threadIdx.x; idx < total;
       idx += (i64)gridDim.x * blockDim.x) {
    const int c = (int)(idx % C);
    const float mean = stats[c] * invM;
    const float var  = stats[C + c] * invM - mean * mean;
    float y = g[c] * (buf[idx] - mean) * rsqrtf(var + 1e-5f) + be[c];
    y = (y >= 0.f) ? y : 0.2f * y;         // LeakyReLU(0.2)
    obf[idx] = f2bf(y);
    if (of32) of32[idx] = y;
  }
}

// ---------------------------------------------------------------------------
__global__ __launch_bounds__(256) void k_row_norm(const float* __restrict__ A,
                                                  float* __restrict__ norms, int K) {
  __shared__ float red[256];
  const int row = blockIdx.x, t = threadIdx.x;
  float s = 0.f;
  for (int j = t; j < K; j += 256) {
    const float v = A[(i64)row * K + j];
    s += v * v;
  }
  red[t] = s; __syncthreads();
  for (int off = 128; off; off >>= 1) {
    if (t < off) red[t] += red[t + off];
    __syncthreads();
  }
  if (t == 0) norms[row] = sqrtf(red[0]);
}

// mem[2000][256] -> B for z@mem^T : out[(k/2)*2000 + n] = pair(mem[n][k..k+1])
__global__ void k_pack_memT(const float* __restrict__ mem, u32* __restrict__ out) {
  const i64 total = 128LL * 2000;
  for (i64 idx = blockIdx.x * (i64)blockDim.x + threadIdx.x; idx < total;
       idx += (i64)gridDim.x * blockDim.x) {
    const int n  = (int)(idx % 2000);
    const int kp = (int)(idx / 2000);
    const float lo = mem[(i64)n * 256 + 2 * kp];
    const float hi = mem[(i64)n * 256 + 2 * kp + 1];
    out[idx] = (u32)f2bf(lo) | ((u32)f2bf(hi) << 16);
  }
}

// mem[2000][256] -> B for w@mem : out[kp*256 + n], K padded to 2016
__global__ void k_pack_memN(const float* __restrict__ mem, u32* __restrict__ out) {
  const i64 total = 1008LL * 256;
  for (i64 idx = blockIdx.x * (i64)blockDim.x + threadIdx.x; idx < total;
       idx += (i64)gridDim.x * blockDim.x) {
    const int n  = (int)(idx % 256);
    const int kp = (int)(idx / 256);
    const int k  = 2 * kp;
    const float lo = (k     < 2000) ? mem[(i64)k * 256 + n]       : 0.f;
    const float hi = (k + 1 < 2000) ? mem[(i64)(k + 1) * 256 + n] : 0.f;
    out[idx] = (u32)f2bf(lo) | ((u32)f2bf(hi) << 16);
  }
}

// ---------------------------------------------------------------------------
// Cosine-sim softmax + ReLU + L1 renorm, one block per row (2000 cols)
// ---------------------------------------------------------------------------
__global__ __launch_bounds__(256) void k_softmax(
    const float* __restrict__ scores, const float* __restrict__ znorm,
    const float* __restrict__ mnorm, float* __restrict__ w_out,
    u16* __restrict__ w_bf) {
  __shared__ float red[256];
  const int row = blockIdx.x, t = threadIdx.x;
  const float zn = znorm[row];
  float s[8];
#pragma unroll
  for (int i = 0; i < 8; ++i) {
    const int j = t + i * 256;
    if (j < 2000) {
      const float den = fmaxf(zn * mnorm[j], 1e-8f);
      s[i] = scores[(i64)row * 2000 + j] / den;
    } else s[i] = -3.0e38f;
  }
  float mx = s[0];
#pragma unroll
  for (int i = 1; i < 8; ++i) mx = fmaxf(mx, s[i]);
  red[t] = mx; __syncthreads();
  for (int off = 128; off; off >>= 1) { if (t < off) red[t] = fmaxf(red[t], red[t + off]); __syncthreads(); }
  mx = red[0]; __syncthreads();

  float sum = 0.f;
#pragma unroll
  for (int i = 0; i < 8; ++i) {
    if (t + i * 256 < 2000) { s[i] = __expf(s[i] - mx); sum += s[i]; }
    else s[i] = 0.f;
  }
  red[t] = sum; __syncthreads();
  for (int off = 128; off; off >>= 1) { if (t < off) red[t] += red[t + off]; __syncthreads(); }
  sum = red[0]; __syncthreads();
  const float inv = 1.0f / sum;

  float l1 = 0.f;
#pragma unroll
  for (int i = 0; i < 8; ++i) { s[i] *= inv; l1 += fabsf(s[i]); }
  red[t] = l1; __syncthreads();
  for (int off = 128; off; off >>= 1) { if (t < off) red[t] += red[t + off]; __syncthreads(); }
  l1 = red[0];
  const float invl1 = 1.0f / fmaxf(l1, 1e-12f);
#pragma unroll
  for (int i = 0; i < 8; ++i) {
    const int j = t + i * 256;
    if (j < 2000) {
      const float wh = s[i] * invl1;
      w_out[(i64)row * 2000 + j] = wh;
      w_bf[(i64)row * 2016 + j] = f2bf(wh);
    }
  }
  if (t < 16) w_bf[(i64)row * 2016 + 2000 + t] = 0;  // K pad for WMMA GEMM
}

// ---------------------------------------------------------------------------
extern "C" void kernel_launch(void* const* d_in, const int* in_sizes, int n_in,
                              void* d_out, int out_size, void* d_ws, size_t ws_size,
                              hipStream_t stream) {
  (void)in_sizes; (void)n_in; (void)out_size; (void)ws_size;

  const float* x = (const float*)d_in[0];
  const float *ew[4], *eb[4], *eg[4], *ebe[4];
  for (int i = 0; i < 4; ++i) {
    ew[i]  = (const float*)d_in[1 + 4 * i];
    eb[i]  = (const float*)d_in[2 + 4 * i];
    eg[i]  = (const float*)d_in[3 + 4 * i];
    ebe[i] = (const float*)d_in[4 + 4 * i];
  }
  const float* mem = (const float*)d_in[17];
  const float *dw[4], *db[4], *dg[3], *dbe[3];
  for (int i = 0; i < 3; ++i) {
    dw[i]  = (const float*)d_in[18 + 4 * i];
    db[i]  = (const float*)d_in[19 + 4 * i];
    dg[i]  = (const float*)d_in[20 + 4 * i];
    dbe[i] = (const float*)d_in[21 + 4 * i];
  }
  dw[3] = (const float*)d_in[30];
  db[3] = (const float*)d_in[31];

  float* outY = (float*)d_out;                 // (2,1,16,256,256)
  float* outW = outY + 2097152;                // (1024,2000)

  // ---- workspace carve ----
  char* ws = (char*)d_ws;
  size_t off = 0;
  auto carve = [&](size_t bytes) -> void* {
    off = (off + 255) & ~(size_t)255;
    void* p = ws + off; off += bytes; return p;
  };
  u16*   xbf    = (u16*)  carve(2097152ULL * 2);
  float* f32buf = (float*)carve(50331648ULL * 4);   // largest M*C fp32
  u16*   bfA    = (u16*)  carve(50331648ULL * 2);
  u16*   bfB    = (u16*)  carve(50331648ULL * 2);
  float* stats  = (float*)carve(512 * 4);
  float* zf32   = (float*)carve(1048576ULL * 4);
  float* znorm  = (float*)carve(1024 * 4);
  float* mnorm  = (float*)carve(2000 * 4);
  float* scores = (float*)carve(1024ULL * 2000 * 4);
  u16*   whatbf = (u16*)  carve(1024ULL * 2016 * 2);
  float* zhat   = (float*)carve(1048576ULL * 4);
  u16*   zeropad= (u16*)  carve(256);
  u32* wp_e1 = (u32*)carve(16ULL   * 96  * 4);
  u32* wp_e2 = (u32*)carve(1296ULL * 128 * 4);
  u32* wp_e3 = (u32*)carve(1728ULL * 256 * 4);
  u32* wp_e4 = (u32*)carve(3456ULL * 256 * 4);
  u32* wp_d1 = (u32*)carve(3456ULL * 256 * 4);
  u32* wp_d2 = (u32*)carve(3456ULL * 128 * 4);
  u32* wp_d3 = (u32*)carve(1728ULL * 96  * 4);
  u32* wp_d4 = (u32*)carve(1296ULL * 16  * 4);
  u32* mpT   = (u32*)carve(128ULL  * 2000 * 4);
  u32* mpN   = (u32*)carve(1008ULL * 256  * 4);

  auto gs = [](i64 total) { i64 b = (total + 255) / 256; return (unsigned)(b < 4096 ? b : 4096); };

  k_zero<<<1, 64, 0, stream>>>((float*)zeropad, 64);

  // ---- weight packing (fragment-major for convs) ----
  k_reorder_w_frag<<<gs(16LL * 96),    256, 0, stream>>>(ew[0], wp_e1, 96, 1,   96,  32,   0);
  k_reorder_w_frag<<<gs(1296LL * 128), 256, 0, stream>>>(ew[1], wp_e2, 128, 96, 128, 2592, 0);
  k_reorder_w_frag<<<gs(1728LL * 256), 256, 0, stream>>>(ew[2], wp_e3, 256, 128,256, 3456, 0);
  k_reorder_w_frag<<<gs(3456LL * 256), 256, 0, stream>>>(ew[3], wp_e4, 256, 256,256, 6912, 0);
  k_reorder_w_frag<<<gs(3456LL * 256), 256, 0, stream>>>(dw[0], wp_d1, 256, 256,256, 6912, 1);
  k_reorder_w_frag<<<gs(3456LL * 128), 256, 0, stream>>>(dw[1], wp_d2, 128, 256,128, 6912, 1);
  k_reorder_w_frag<<<gs(1728LL * 96),  256, 0, stream>>>(dw[2], wp_d3, 96, 128, 96,  3456, 1);
  k_reorder_w_frag<<<gs(1296LL * 16),  256, 0, stream>>>(dw[3], wp_d4, 1,  96,  16,  2592, 1);
  k_pack_memT<<<gs(128LL * 2000), 256, 0, stream>>>(mem, mpT);
  k_pack_memN<<<gs(1008LL * 256), 256, 0, stream>>>(mem, mpN);
  k_row_norm<<<2000, 256, 0, stream>>>(mem, mnorm, 256);

  k_f32_to_bf16<<<gs(2097152), 256, 0, stream>>>(x, xbf, 2097152);

  auto runConv = [&](const u16* act, const u32* wp, const float* bias, float* out_,
                     int Din, int Hin, int Win, int Cin,
                     int Dout, int Hout, int Wout, int CoutPad, int nStore,
                     int sd, int sh, int sw, int dd, int dh, int dw_,
                     int Ktot, int Kpad) {
    const i64 M = 2LL * Dout * Hout * Wout;
    dim3 g((unsigned)(M / 32), (unsigned)((CoutPad + 127) / 128));
    k_conv_wmma<<<g, 256, 0, stream>>>(act, wp, bias, out_, zeropad,
        Din, Hin, Win, Cin, Dout, Hout, Wout, CoutPad, nStore,
        sd, sh, sw, dd, dh, dw_, Ktot, Kpad);
  };
  auto runBN = [&](const float* buf, const float* g, const float* be,
                   u16* obf, float* of32, i64 M, int C) {
    k_zero<<<2, 256, 0, stream>>>(stats, 2 * C);
    k_bn_stats<<<512, C, 0, stream>>>(buf, stats, M, C);
    k_bn_apply<<<gs(M * C), 256, 0, stream>>>(buf, stats, g, be, obf, of32, M, C);
  };

  // ---- encoder ----
  runConv(xbf, wp_e1, eb[0], f32buf, 16,256,256, 1,   16,128,128, 96, 96,  1,2,2, 1,1,1, 27,   32);
  runBN(f32buf, eg[0], ebe[0], bfA, nullptr, 524288, 96);
  runConv(bfA, wp_e2, eb[1], f32buf, 16,128,128, 96, 8,64,64,  128,128, 2,2,2, 1,1,1, 2592, 2592);
  runBN(f32buf, eg[1], ebe[1], bfB, nullptr, 65536, 128);
  runConv(bfB, wp_e3, eb[2], f32buf, 8,64,64, 128,  4,32,32,  256,256, 2,2,2, 1,1,1, 3456, 3456);
  runBN(f32buf, eg[2], ebe[2], bfA, nullptr, 8192, 256);
  runConv(bfA, wp_e4, eb[3], f32buf, 4,32,32, 256,  2,16,16,  256,256, 2,2,2, 1,1,1, 6912, 6912);
  runBN(f32buf, eg[3], ebe[3], bfB, zf32, 1024, 256);

  // ---- memory addressing ----
  k_row_norm<<<1024, 256, 0, stream>>>(zf32, znorm, 256);
  k_gemm_wmma<<<dim3(64, 16), 256, 0, stream>>>(bfB, mpT, scores, 2000, 256);
  k_softmax<<<1024, 256, 0, stream>>>(scores, znorm, mnorm, outW, whatbf);
  k_gemm_wmma<<<dim3(64, 2), 256, 0, stream>>>(whatbf, mpN, zhat, 256, 2016);
  k_f32_to_bf16<<<gs(262144), 256, 0, stream>>>(zhat, bfA, 262144);

  // ---- decoder (convT: stride->lhs dilation, weights pre-flipped) ----
  runConv(bfA, wp_d1, db[0], f32buf, 2,16,16, 256,   4,32,32,   256,256, 1,1,1, 2,2,2, 6912, 6912);
  runBN(f32buf, dg[0], dbe[0], bfB, nullptr, 8192, 256);
  runConv(bfB, wp_d2, db[1], f32buf, 4,32,32, 256,   8,64,64,   128,128, 1,1,1, 2,2,2, 6912, 6912);
  runBN(f32buf, dg[1], dbe[1], bfA, nullptr, 65536, 128);
  runConv(bfA, wp_d3, db[2], f32buf, 8,64,64, 128,  16,128,128, 96, 96,  1,1,1, 2,2,2, 3456, 3456);
  runBN(f32buf, dg[2], dbe[2], bfB, nullptr, 524288, 96);
  runConv(bfB, wp_d4, db[3], outY,   16,128,128, 96, 16,256,256, 16, 1,  1,1,1, 1,2,2, 2592, 2592);
}
